// PDETransolver_34479997452564
// MI455X (gfx1250) — compile-verified, matrix operand order checked by reading, not run on hardware
//
#include <hip/hip_runtime.h>
#include <hip/hip_bf16.h>

// ---------------------------------------------------------------------------
// PDE-Transolver forward for MI455X (gfx1250, wave32, WMMA).
// ~180 GFLOP of GEMMs vs ~0.5 GB activation traffic -> compute bound; all
// matmuls routed through v_wmma_f32_16x16x32_bf16 (f32->bf16 truncate, f32 acc).
// LDS tiles are stored pre-permuted into the ISA 16-bit fragment order so each
// lane's fragment is one aligned 32B vector load (ds_load_b128 x2), not a
// scalar ds_load_u16 gather.
// ---------------------------------------------------------------------------

typedef __attribute__((ext_vector_type(16))) __bf16 v16bf;
typedef __attribute__((ext_vector_type(8)))  __bf16 v8bf;
typedef __attribute__((ext_vector_type(8)))  float  v8f;

#define B_    2
#define T_    8
#define N_    1024      // patches per frame
#define D_    256
#define S_    64        // slice tokens
#define NH_   8
#define HD_   32
#define MLPH_ 1024
#define BT_   (B_ * T_)         // 16
#define ROWS_ (BT_ * N_)        // 16384 token rows

__device__ __forceinline__ __bf16 f2bf(float f) {
  unsigned u = __builtin_bit_cast(unsigned, f);
  unsigned short h = (unsigned short)(u >> 16);
  return __builtin_bit_cast(__bf16, h);
}

// ---------------------------------------------------------------------------
// Batched GEMM: C[b] = act(A[b] @ B[b] + bias) (+ Res[b])
// A: [M,K] f32 (or [K,M] when transA), B: [K,N] f32, C: [M,N] f32.
// Requires M%64==0, N%64==0, K%32==0 (true for every call site here).
// Block: 128 threads = 4 waves; block tile 64x64; wave tile 16x64.
//
// LDS layout: As[row][pos], Bs[col][pos] with pos = fragment-order permutation
// of k (groups of 8: k0-7 -> 0-7, k8-15 -> 16-23, k16-23 -> 8-15, k24-31 ->
// 24-31). Lane group g (lane>>4) reads 16 contiguous bf16 at pos = g*16.
// Row pitch 48 elems (96B) keeps every fragment address 32B-aligned.
// ---------------------------------------------------------------------------
__global__ __launch_bounds__(128)
void gemm_bf16_wmma(const float* __restrict__ A, long long sA, int lda, int transA,
                    const float* __restrict__ Bm, long long sB, int ldb,
                    const float* __restrict__ bias,
                    const float* __restrict__ Res, long long sRes,
                    float* __restrict__ C, long long sC, int ldc,
                    int M, int N, int K, int act)
{
  __shared__ __align__(64) __bf16 As[64][48];
  __shared__ __align__(64) __bf16 Bs[64][48];

  const int bz = blockIdx.z;
  A  += (long long)bz * sA;
  Bm += (long long)bz * sB;
  C  += (long long)bz * sC;
  if (Res) Res += (long long)bz * sRes;

  const int m0 = blockIdx.y * 64;
  const int n0 = blockIdx.x * 64;
  const int tid  = threadIdx.x;
  const int wave = tid >> 5;
  const int lane = tid & 31;

  // Staging role: this thread owns A-row / B-col `srow`, K-half `shalf`.
  const int srow  = tid & 63;
  const int shalf = tid >> 6;          // 0: k 0..15, 1: k 16..31
  // Permuted store bases: j=0..7 -> 16*? see header comment.
  const int pos0 = shalf * 8;          // k j=0..7  of this half
  const int pos1 = 16 + shalf * 8;     // k j=8..15 of this half

  v8f acc[4] = {};

  for (int kk = 0; kk < K; kk += 32) {
    // ---- stage A (64x32) ----
    float a[16];
    if (transA) {
      #pragma unroll
      for (int j = 0; j < 16; ++j)
        a[j] = A[(long long)(kk + shalf * 16 + j) * lda + (m0 + srow)];
    } else {
      const float* ap = A + (long long)(m0 + srow) * lda + kk + shalf * 16;
      #pragma unroll
      for (int j = 0; j < 16; ++j) a[j] = ap[j];
    }
    v8bf ac0, ac1;
    #pragma unroll
    for (int j = 0; j < 8; ++j) { ac0[j] = f2bf(a[j]); ac1[j] = f2bf(a[8 + j]); }
    *(v8bf*)&As[srow][pos0] = ac0;
    *(v8bf*)&As[srow][pos1] = ac1;

    // ---- stage B (32x64), column-walk (coalesced across lanes per k) ----
    float b[16];
    #pragma unroll
    for (int j = 0; j < 16; ++j)
      b[j] = Bm[(long long)(kk + shalf * 16 + j) * ldb + (n0 + srow)];
    v8bf bc0, bc1;
    #pragma unroll
    for (int j = 0; j < 8; ++j) { bc0[j] = f2bf(b[j]); bc1[j] = f2bf(b[8 + j]); }
    *(v8bf*)&Bs[srow][pos0] = bc0;
    *(v8bf*)&Bs[srow][pos1] = bc1;

    __syncthreads();

    // ---- fragments: one aligned 32B vector load each ----
    const int fr = lane & 15;
    const int fo = (lane >> 4) * 16;
    const v16bf af = *(const v16bf*)&As[wave * 16 + fr][fo];
    #pragma unroll
    for (int nt = 0; nt < 4; ++nt) {
      const v16bf bfv = *(const v16bf*)&Bs[nt * 16 + fr][fo];
      acc[nt] = __builtin_amdgcn_wmma_f32_16x16x32_bf16(
          false, af, false, bfv, (short)0, acc[nt], false, false);
    }
    __syncthreads();
  }

  // Epilogue. C/D layout: VGPR r -> M = r + 8*(lane>=16), N = lane&15.
  const int grp  = lane >> 4;
  const int colL = lane & 15;
  #pragma unroll
  for (int nt = 0; nt < 4; ++nt) {
    const int col = n0 + nt * 16 + colL;
    const float bia = bias ? bias[col] : 0.0f;
    #pragma unroll
    for (int r = 0; r < 8; ++r) {
      const int row = m0 + wave * 16 + grp * 8 + r;
      float v = acc[nt][r] + bia;
      if (act == 1) v = 0.5f * v * (1.0f + erff(v * 0.70710678118f));
      if (Res) v += Res[(long long)row * ldc + col];
      C[(long long)row * ldc + col] = v;
    }
  }
}

// ---------------------------------------------------------------------------
// LayerNorm over D=256; optional pre-scale by 1/(sums[row]+1e-6) (token norm).
// ---------------------------------------------------------------------------
__global__ __launch_bounds__(256)
void ln_kernel(const float* __restrict__ x, const float* __restrict__ sums,
               const float* __restrict__ w, const float* __restrict__ bb,
               float* __restrict__ y)
{
  __shared__ float red[256];
  const int row = blockIdx.x, t = threadIdx.x;
  const long long base = (long long)row * D_;
  float v = x[base + t];
  if (sums) v = v / (sums[row] + 1e-6f);
  red[t] = v; __syncthreads();
  #pragma unroll
  for (int s = 128; s > 0; s >>= 1) { if (t < s) red[t] += red[t + s]; __syncthreads(); }
  const float mean = red[0] * (1.0f / D_);
  __syncthreads();
  const float d = v - mean;
  red[t] = d * d; __syncthreads();
  #pragma unroll
  for (int s = 128; s > 0; s >>= 1) { if (t < s) red[t] += red[t + s]; __syncthreads(); }
  const float var = red[0] * (1.0f / D_);
  y[base + t] = d * rsqrtf(var + 1e-5f) * w[t] + bb[t];
}

// Softmax over S=64 slice logits with clipped temperature (in place).
__global__ __launch_bounds__(64)
void softmax_slice_kernel(float* __restrict__ sw, const float* __restrict__ temp)
{
  __shared__ float red[64];
  const int row = blockIdx.x, t = threadIdx.x;
  const float tp = fmaxf(temp[0], 0.1f);
  const float v = sw[(long long)row * S_ + t] / tp;
  red[t] = v; __syncthreads();
  #pragma unroll
  for (int s = 32; s > 0; s >>= 1) { if (t < s) red[t] = fmaxf(red[t], red[t + s]); __syncthreads(); }
  const float mx = red[0]; __syncthreads();
  const float e = expf(v - mx);
  red[t] = e; __syncthreads();
  #pragma unroll
  for (int s = 32; s > 0; s >>= 1) { if (t < s) red[t] += red[t + s]; __syncthreads(); }
  sw[(long long)row * S_ + t] = e / red[0];
}

// sums[bt*S + s] = sum_n sw[bt, n, s]
__global__ __launch_bounds__(256)
void colsum_kernel(const float* __restrict__ sw, float* __restrict__ sums)
{
  const int bt = blockIdx.x >> 6, s = blockIdx.x & 63;
  const float* p = sw + (long long)bt * N_ * S_ + s;
  float acc = 0.0f;
  for (int n = threadIdx.x; n < N_; n += 256) acc += p[(long long)n * S_];
  __shared__ float red[256];
  red[threadIdx.x] = acc; __syncthreads();
  #pragma unroll
  for (int st = 128; st > 0; st >>= 1) {
    if (threadIdx.x < st) red[threadIdx.x] += red[threadIdx.x + st];
    __syncthreads();
  }
  if (threadIdx.x == 0) sums[blockIdx.x] = red[0];
}

// Slice-token attention: one block per (bt, head); S=64, HD=32.
__global__ __launch_bounds__(256)
void attention_kernel(const float* __restrict__ q, const float* __restrict__ k,
                      const float* __restrict__ v, float* __restrict__ o)
{
  __shared__ float qs[S_][HD_], ks[S_][HD_], vs[S_][HD_], sc[S_][S_];
  const int bt = blockIdx.x >> 3, h = blockIdx.x & 7;
  const long long base = (long long)bt * S_ * D_ + h * HD_;
  const int t = threadIdx.x;
  for (int i = t; i < S_ * HD_; i += 256) {
    const int s = i >> 5, d = i & 31;
    qs[s][d] = q[base + (long long)s * D_ + d];
    ks[s][d] = k[base + (long long)s * D_ + d];
    vs[s][d] = v[base + (long long)s * D_ + d];
  }
  __syncthreads();
  const float scale = 0.1767766953f; // HD^-0.5
  for (int i = t; i < S_ * S_; i += 256) {
    const int s = i >> 6, u = i & 63;
    float a = 0.0f;
    #pragma unroll
    for (int d = 0; d < HD_; ++d) a += qs[s][d] * ks[u][d];
    sc[s][u] = a * scale;
  }
  __syncthreads();
  if (t < S_) {
    float mx = -1e30f;
    for (int u = 0; u < S_; ++u) mx = fmaxf(mx, sc[t][u]);
    float sum = 0.0f;
    for (int u = 0; u < S_; ++u) { const float e = expf(sc[t][u] - mx); sc[t][u] = e; sum += e; }
    const float inv = 1.0f / sum;
    for (int u = 0; u < S_; ++u) sc[t][u] *= inv;
  }
  __syncthreads();
  for (int i = t; i < S_ * HD_; i += 256) {
    const int s = i >> 5, d = i & 31;
    float a = 0.0f;
    for (int u = 0; u < S_; ++u) a += sc[s][u] * vs[u][d];
    o[base + (long long)s * D_ + d] = a;
  }
}

// Causal depthwise temporal conv (k=3) + LN + residual: out = h + LN(conv(h)).
__global__ __launch_bounds__(256)
void tconv_kernel(const float* __restrict__ h, const float* __restrict__ tc_w,
                  const float* __restrict__ tc_b, const float* __restrict__ tn_w,
                  const float* __restrict__ tn_b, float* __restrict__ out)
{
  const int idx = blockIdx.x;                 // b*T*N + t*N + n
  const int c = threadIdx.x;
  const int tt = (idx / N_) % T_;
  const long long rbase = (long long)idx * D_;
  const long long strT = (long long)N_ * D_;
  const float x0 = (tt >= 2) ? h[rbase - 2 * strT + c] : 0.0f;
  const float x1 = (tt >= 1) ? h[rbase - strT + c] : 0.0f;
  const float x2 = h[rbase + c];
  const float conv = x0 * tc_w[c * 3 + 0] + x1 * tc_w[c * 3 + 1]
                   + x2 * tc_w[c * 3 + 2] + tc_b[c];
  __shared__ float red[256];
  red[c] = conv; __syncthreads();
  #pragma unroll
  for (int s = 128; s > 0; s >>= 1) { if (c < s) red[c] += red[c + s]; __syncthreads(); }
  const float mean = red[0] * (1.0f / D_);
  __syncthreads();
  const float d = conv - mean;
  red[c] = d * d; __syncthreads();
  #pragma unroll
  for (int s = 128; s > 0; s >>= 1) { if (c < s) red[c] += red[c + s]; __syncthreads(); }
  const float var = red[0] * (1.0f / D_);
  out[rbase + c] = x2 + d * rsqrtf(var + 1e-5f) * tn_w[c] + tn_b[c];
}

// Patchify: xp[bt, n, f] with f = c*16 + pr*4 + pc.
__global__ __launch_bounds__(256)
void im2col_kernel(const float* __restrict__ x, float* __restrict__ xp)
{
  const long long i = (long long)blockIdx.x * 256 + threadIdx.x; // 16*1024*64
  const int f = (int)(i & 63);
  const long long rest = i >> 6;
  const int n = (int)(rest & 1023);
  const int bt = (int)(rest >> 10);
  const int c = f >> 4, pr = (f >> 2) & 3, pc = f & 3;
  const int hi = (n >> 5) * 4 + pr, wi = (n & 31) * 4 + pc;
  xp[i] = x[(((long long)bt * 128 + hi) * 128 + wi) * 4 + c];
}

__global__ __launch_bounds__(256)
void addpos_kernel(float* __restrict__ h, const float* __restrict__ pos)
{
  const long long i = (long long)blockIdx.x * 256 + threadIdx.x;
  const int d = (int)(i & 255);
  const int n = (int)((i >> 8) & 1023);
  h[i] += pos[(long long)n * D_ + d];
}

// Fold [bt,n,64] head output back to image + 1x1 residual projection + input.
__global__ __launch_bounds__(256)
void fold_res_kernel(const float* __restrict__ x, const float* __restrict__ y,
                     const float* __restrict__ res_w, const float* __restrict__ res_b,
                     float* __restrict__ out)
{
  const long long i = (long long)blockIdx.x * 256 + threadIdx.x; // B*T*128*128*4
  const int c = (int)(i & 3);
  long long r = i >> 2;
  const int wi = (int)(r & 127); r >>= 7;
  const int hi = (int)(r & 127); r >>= 7;
  const int bt = (int)r;
  const int n = (hi >> 2) * 32 + (wi >> 2);
  const int pr = hi & 3, pc = wi & 3;
  const float* hrow = y + ((long long)bt * N_ + n) * 64;
  float acc = res_b[c];
  #pragma unroll
  for (int cp = 0; cp < 4; ++cp) acc += hrow[cp * 16 + pr * 4 + pc] * res_w[cp * 4 + c];
  out[i] = x[i] + acc;
}

// ---------------------------------------------------------------------------
// Host side
// ---------------------------------------------------------------------------
static inline void gemm(hipStream_t st,
                        const float* A, long long sA, int lda, int transA,
                        const float* Bm, long long sB, int ldb,
                        const float* bias, const float* Res, long long sRes,
                        float* C, long long sC, int ldc,
                        int M, int N, int K, int act, int batch)
{
  dim3 g(N / 64, M / 64, batch);
  gemm_bf16_wmma<<<g, 128, 0, st>>>(A, sA, lda, transA, Bm, sB, ldb,
                                    bias, Res, sRes, C, sC, ldc, M, N, K, act);
}

extern "C" void kernel_launch(void* const* d_in, const int* in_sizes, int n_in,
                              void* d_out, int out_size, void* d_ws, size_t ws_size,
                              hipStream_t stream)
{
  (void)in_sizes; (void)n_in; (void)out_size; (void)ws_size;
  auto F = [&](int i) { return (const float*)d_in[i]; };

  // Pytree leaf order: top-level dict insertion order {x, params}; params and
  // block dicts flattened alphabetically (jax tree_leaves convention).
  const float* x = F(0);
  const int BLK0 = 1, BLKSTRIDE = 25;           // blocks[0..7], 25 leaves each
  const int PB = 1 + 8 * BLKSTRIDE;             // = 201
  const float* norm_b  = F(PB + 0);
  const float* norm_w  = F(PB + 1);
  const float* out1_b  = F(PB + 2);
  const float* out1_w  = F(PB + 3);
  const float* out2_b  = F(PB + 4);
  const float* out2_w  = F(PB + 5);
  const float* patch_b = F(PB + 6);
  const float* patch_w = F(PB + 7);
  const float* pos     = F(PB + 8);
  const float* res_b   = F(PB + 9);
  const float* res_w   = F(PB + 10);

  // Workspace arena (f32).
  float* W = (float*)d_ws;
  size_t off = 0;
  auto alloc = [&](size_t n) { float* p = W + off; off += n; return p; };
  float* hA   = alloc((size_t)ROWS_ * D_);
  float* hB   = alloc((size_t)ROWS_ * D_);
  float* lnb  = alloc((size_t)ROWS_ * D_);
  float* dsb  = alloc((size_t)ROWS_ * D_);        // deslice result / head output
  float* mlp  = alloc((size_t)ROWS_ * MLPH_);     // also im2col + out1 staging
  float* sw   = alloc((size_t)ROWS_ * S_);
  float* tok  = alloc((size_t)BT_ * S_ * D_);
  float* tokn = alloc((size_t)BT_ * S_ * D_);
  float* qb   = alloc((size_t)BT_ * S_ * D_);
  float* kb   = alloc((size_t)BT_ * S_ * D_);
  float* vb   = alloc((size_t)BT_ * S_ * D_);
  float* ob   = alloc((size_t)BT_ * S_ * D_);
  float* sums = alloc((size_t)BT_ * S_);

  // ---- patch embed: h = im2col(x) @ patch_w + patch_b + pos
  im2col_kernel<<<4096, 256, 0, stream>>>(x, mlp);
  gemm(stream, mlp, 0, 64, 0, patch_w, 0, D_, patch_b, nullptr, 0,
       hA, 0, D_, ROWS_, D_, 64, 0, 1);
  addpos_kernel<<<ROWS_, 256, 0, stream>>>(hA, pos);

  float* hcur = hA;
  float* halt = hB;

  for (int l = 0; l < 8; ++l) {
    const int b0 = BLK0 + l * BLKSTRIDE;
    const float *k_b = F(b0+0),  *k_w = F(b0+1),  *m1_b = F(b0+2), *m1_w = F(b0+3);
    const float *m2_b = F(b0+4), *m2_w = F(b0+5), *n1_b = F(b0+6), *n1_w = F(b0+7);
    const float *n2_b = F(b0+8), *n2_w = F(b0+9), *o_b = F(b0+10), *o_w = F(b0+11);
    const float *q_b = F(b0+12), *q_w = F(b0+13), *slice_b = F(b0+14), *slice_w = F(b0+15);
    const float *sn_b = F(b0+16),*sn_w = F(b0+17),*tc_b = F(b0+18), *tc_w = F(b0+19);
    const float *temp = F(b0+20),*tn_b = F(b0+21),*tn_w = F(b0+22);
    const float *v_b = F(b0+23), *v_w = F(b0+24);

    // ln1 = LN(h)
    ln_kernel<<<ROWS_, 256, 0, stream>>>(hcur, nullptr, n1_w, n1_b, lnb);
    // sw = softmax((ln1 @ slice_w + slice_b) / temp)
    gemm(stream, lnb, 0, D_, 0, slice_w, 0, S_, slice_b, nullptr, 0,
         sw, 0, S_, ROWS_, S_, D_, 0, 1);
    softmax_slice_kernel<<<ROWS_, 64, 0, stream>>>(sw, temp);
    colsum_kernel<<<BT_ * S_, 256, 0, stream>>>(sw, sums);
    // tok[bt] = sw[bt]^T @ h[bt]   (M=S, K=N, N=D; batched over bt)
    gemm(stream, sw, (long long)N_ * S_, S_, 1,
         hcur, (long long)N_ * D_, D_, nullptr, nullptr, 0,
         tok, (long long)S_ * D_, D_, S_, D_, N_, 0, BT_);
    // tok /= (colsum + 1e-6); LN(sn)
    ln_kernel<<<BT_ * S_, 256, 0, stream>>>(tok, sums, sn_w, sn_b, tokn);
    // q/k/v projections (M = BT*S = 1024)
    gemm(stream, tokn, 0, D_, 0, q_w, 0, D_, q_b, nullptr, 0, qb, 0, D_, BT_*S_, D_, D_, 0, 1);
    gemm(stream, tokn, 0, D_, 0, k_w, 0, D_, k_b, nullptr, 0, kb, 0, D_, BT_*S_, D_, D_, 0, 1);
    gemm(stream, tokn, 0, D_, 0, v_w, 0, D_, v_b, nullptr, 0, vb, 0, D_, BT_*S_, D_, D_, 0, 1);
    attention_kernel<<<BT_ * NH_, 256, 0, stream>>>(qb, kb, vb, ob);
    // deslice: ds[bt] = sw[bt] @ o[bt]  (M=N, K=S, N=D; batched)
    gemm(stream, sw, (long long)N_ * S_, S_, 0,
         ob, (long long)S_ * D_, D_, nullptr, nullptr, 0,
         dsb, (long long)N_ * D_, D_, N_, D_, S_, 0, BT_);
    // h = h + ds @ o_w + o_b (residual fused in epilogue, in place)
    gemm(stream, dsb, 0, D_, 0, o_w, 0, D_, o_b, hcur, 0,
         hcur, 0, D_, ROWS_, D_, D_, 0, 1);
    // MLP: h = h + gelu(LN(h) @ m1_w + m1_b) @ m2_w + m2_b
    ln_kernel<<<ROWS_, 256, 0, stream>>>(hcur, nullptr, n2_w, n2_b, lnb);
    gemm(stream, lnb, 0, D_, 0, m1_w, 0, MLPH_, m1_b, nullptr, 0,
         mlp, 0, MLPH_, ROWS_, MLPH_, D_, 1, 1);
    gemm(stream, mlp, 0, MLPH_, 0, m2_w, 0, D_, m2_b, hcur, 0,
         hcur, 0, D_, ROWS_, D_, MLPH_, 0, 1);
    // temporal depthwise conv + LN + residual (ping-pong to avoid RAW over T)
    tconv_kernel<<<ROWS_, 256, 0, stream>>>(hcur, tc_w, tc_b, tn_w, tn_b, halt);
    float* tswap = hcur; hcur = halt; halt = tswap;
  }

  // Head: LN -> gelu(out1) -> out2 -> fold + 1x1 residual projection.
  ln_kernel<<<ROWS_, 256, 0, stream>>>(hcur, nullptr, norm_w, norm_b, lnb);
  gemm(stream, lnb, 0, D_, 0, out1_w, 0, 2 * D_, out1_b, nullptr, 0,
       mlp, 0, 2 * D_, ROWS_, 2 * D_, D_, 1, 1);
  gemm(stream, mlp, 0, 2 * D_, 0, out2_w, 0, 64, out2_b, nullptr, 0,
       dsb, 0, 64, ROWS_, 64, 2 * D_, 0, 1);
  fold_res_kernel<<<4096, 256, 0, stream>>>(x, dsb, res_w, res_b, (float*)d_out);
}
